// SenEncodeModel_60962765799900
// MI455X (gfx1250) — matrix operand order repaired and checked by reading, not compiled, not run on hardware
//
#include <hip/hip_runtime.h>
#include <hip/hip_bf16.h>

// ---------------------------------------------------------------------------
// Maxout-LSTM encoder for MI455X (gfx1250, wave32, WMMA).
//
// Strategy:
//   1) Precompute S_pre[t,b,:] = embed(seq)[b,t] @ Wi + bi + bh  as ONE big
//      bf16 WMMA GEMM (no sequential dependency).  Wave computes 32x32
//      (4 accumulators) -> 4 WMMAs per 8 fragment loads.
//   2) 40 sequential step kernels: s = S_pre[t] + h @ Wh  (bf16 WMMA, fp32
//      accum).  Wave computes M=32 across all 5 gates (10 accumulators,
//      shared A fragments) -> 10 WMMAs per 14 fragment loads.  LSTM maxout
//      epilogue fused in registers; h ping-pongs between bf16 buffers.
//   3) Output written when idx[b] == t (exactly once per batch row).
// ---------------------------------------------------------------------------

#define BN   1024          // batch
#define TN   41            // seq length
#define SN   40            // recurrent steps (T-1)
#define EN   512           // embed dim
#define RN   512           // hidden dim
#define FIVR 2560          // 5*R
#define GN   (SN * BN)     // rows of the big GEMM = 40960

typedef __attribute__((ext_vector_type(16))) __bf16        v16bf;
typedef __attribute__((ext_vector_type(8)))  float         v8f;
typedef __attribute__((ext_vector_type(4)))  unsigned int  v4u;
typedef __attribute__((ext_vector_type(4)))  float         v4f;

union FragA { v16bf v; v4u u[2]; };
union FragB { v16bf v; v4u u[2]; };

__device__ __forceinline__ unsigned short f2bf(float f) {
    unsigned int x = __float_as_uint(f);
    unsigned int r = x + 0x7FFFu + ((x >> 16) & 1u);   // round-to-nearest-even
    return (unsigned short)(r >> 16);
}
__device__ __forceinline__ float sigmoidf_fast(float x) {
    return 1.0f / (1.0f + __expf(-x));
}
__device__ __forceinline__ float tanhf_fast(float x) {
    float e = __expf(2.0f * x);
    return 1.0f - 2.0f / (e + 1.0f);    // saturates cleanly for |x| large
}

// ---------------------------------------------------------------------------
// idx[b] = min(sum(seq_masks[b,:]) - 1, T-2)
// ---------------------------------------------------------------------------
__global__ void idx_kernel(const float* __restrict__ seq_masks, int* __restrict__ idx) {
    int b = blockIdx.x * blockDim.x + threadIdx.x;
    if (b >= BN) return;
    float s = 0.0f;
    #pragma unroll
    for (int t = 0; t < TN; ++t) s += seq_masks[b * TN + t];
    int stop = (int)(s - 0.5f);                 // == (int)s - 1 for integral s
    idx[b] = stop < (TN - 2) ? stop : (TN - 2);
}

// ---------------------------------------------------------------------------
// WT[n,k] = bf16(W[k,n]);  W: [K,N] row-major fp32  ->  WT: [N,K] bf16
// ---------------------------------------------------------------------------
__global__ void transpose_bf_kernel(const float* __restrict__ W,
                                    unsigned short* __restrict__ WT,
                                    int K, int N) {
    long tid = (long)blockIdx.x * blockDim.x + threadIdx.x;
    if (tid >= (long)K * N) return;
    int  k = (int)(tid % K);
    long n = tid / K;
    WT[n * K + k] = f2bf(W[(long)k * N + n]);
}

// ---------------------------------------------------------------------------
// Zero h0 (bf16) and c (fp32), both [B,R]
// ---------------------------------------------------------------------------
__global__ void init_state_kernel(unsigned short* __restrict__ h0,
                                  float* __restrict__ c) {
    int i = blockIdx.x * blockDim.x + threadIdx.x;
    if (i < BN * RN) { h0[i] = 0; c[i] = 0.0f; }
}

// ---------------------------------------------------------------------------
// Xbf[t*B+b, e] = bf16(embed_w[seq[b,t], e]);  8 elements per thread.
// ---------------------------------------------------------------------------
__global__ void embed_gather_kernel(const int* __restrict__ seq,
                                    const float* __restrict__ embed_w,
                                    unsigned short* __restrict__ Xbf) {
    int tid = blockIdx.x * blockDim.x + threadIdx.x;   // GN*EN/8 threads
    if (tid >= GN * EN / 8) return;
    int  e8 = tid & (EN / 8 - 1);      // 0..63
    int  g  = tid >> 6;                // row: t*B + b
    int  b  = g & (BN - 1);
    int  t  = g >> 10;
    int  tok = seq[b * TN + t];
    const float* src = embed_w + (long)tok * EN + e8 * 8;
    v4f lo = *(const v4f*)(src);
    v4f hi = *(const v4f*)(src + 4);
    v4u p;
    p.x = (unsigned int)f2bf(lo.x) | ((unsigned int)f2bf(lo.y) << 16);
    p.y = (unsigned int)f2bf(lo.z) | ((unsigned int)f2bf(lo.w) << 16);
    p.z = (unsigned int)f2bf(hi.x) | ((unsigned int)f2bf(hi.y) << 16);
    p.w = (unsigned int)f2bf(hi.z) | ((unsigned int)f2bf(hi.w) << 16);
    ((v4u*)Xbf)[tid] = p;
}

// ---------------------------------------------------------------------------
// Fragment loads per CDNA5 16-bit WMMA layouts (CDNA5 ISA §7.12.2):
//   A 16x32: lane l holds row (l&15); K = (l>=16? 8:0)+0..7 in v[0..7],
//            +16 in v[8..15]   -> two 16B loads at kh and kh+16
//   B 32x16: lane l holds col (l&15); K = (l>=16? 8:0)+0..7 in v[0..7],
//            +16 in v[8..15]   -> two 16B loads at kh and kh+16
// ---------------------------------------------------------------------------
__device__ __forceinline__ void load_fragA(FragA& a, const unsigned short* base,
                                           int row, int k0, int lane) {
    const unsigned short* p = base + (long)row * EN + k0 + (((lane >> 4) & 1) << 3);
    a.u[0] = *(const v4u*)(p);
    a.u[1] = *(const v4u*)(p + 16);
}
__device__ __forceinline__ void load_fragB(FragB& b, const unsigned short* base,
                                           int col, int k0, int lane) {
    const unsigned short* p = base + (long)col * EN + k0 + (((lane >> 4) & 1) << 3);
    b.u[0] = *(const v4u*)(p);
    b.u[1] = *(const v4u*)(p + 16);
}

#define WMMA_BF16(A, B, C) \
    __builtin_amdgcn_wmma_f32_16x16x32_bf16(false, (A), false, (B), (short)0, (C), false, false)

// ---------------------------------------------------------------------------
// GEMM1: S_pre[g, n] = sum_k Xbf[g,k]*WiT[n,k] + bi[n] + bh[n]
// Block = 256 thr = 8 waves arranged 2(M) x 4(N); wave tile 32x32
// (2x2 accumulators).  Grid: (GN/64, FIVR/128) = (640, 20)
// ---------------------------------------------------------------------------
__global__ __launch_bounds__(256)
void gemm1_kernel(const unsigned short* __restrict__ Xbf,
                  const unsigned short* __restrict__ WiT,
                  const float* __restrict__ bi,
                  const float* __restrict__ bh,
                  float* __restrict__ S_pre) {
    const int wave = threadIdx.x >> 5;
    const int lane = threadIdx.x & 31;
    const int m0 = blockIdx.x * 64  + (wave >> 2) * 32;   // 2 waves along M
    const int n0 = blockIdx.y * 128 + (wave & 3) * 32;    // 4 waves along N
    const int r  = lane & 15;

    v8f acc[2][2] = {};
    for (int k0 = 0; k0 < EN; k0 += 32) {
        FragA a0, a1; FragB b0, b1;
        load_fragA(a0, Xbf, m0 + r,      k0, lane);
        load_fragA(a1, Xbf, m0 + 16 + r, k0, lane);
        load_fragB(b0, WiT, n0 + r,      k0, lane);
        load_fragB(b1, WiT, n0 + 16 + r, k0, lane);
        acc[0][0] = WMMA_BF16(a0.v, b0.v, acc[0][0]);
        acc[0][1] = WMMA_BF16(a0.v, b1.v, acc[0][1]);
        acc[1][0] = WMMA_BF16(a1.v, b0.v, acc[1][0]);
        acc[1][1] = WMMA_BF16(a1.v, b1.v, acc[1][1]);
    }
    const int hi8 = (((lane >> 4) & 1) << 3);
    #pragma unroll
    for (int nt = 0; nt < 2; ++nt) {
        const int n = n0 + nt * 16 + (lane & 15);
        const float bias = bi[n] + bh[n];
        #pragma unroll
        for (int mt = 0; mt < 2; ++mt) {
            const int mbase = m0 + mt * 16 + hi8;
            #pragma unroll
            for (int v = 0; v < 8; ++v)
                S_pre[(long)(mbase + v) * FIVR + n] = acc[mt][nt][v] + bias;
        }
    }
}

// ---------------------------------------------------------------------------
// One recurrent step:
//   acc[mt][g] = h_in(32 rows) @ WhT[g*R + n-tile]   (5 gates, shared A)
//   s          = acc + S_pre_t  -> LSTM maxout epilogue in registers.
// Block = 256 thr = 8 waves arranged 4(M) x 2(N); wave tile M=32,N=16;
// block tile 128x32.  Grid: (B/128, R/32) = (8, 16)
// ---------------------------------------------------------------------------
__global__ __launch_bounds__(256)
void lstm_step_kernel(const unsigned short* __restrict__ h_in,   // [B,R] bf16
                      const unsigned short* __restrict__ WhT,    // [5R,R] bf16
                      const float* __restrict__ S_pre_t,         // [B,5R] fp32
                      float* __restrict__ c,                     // [B,R] fp32
                      unsigned short* __restrict__ h_out,        // [B,R] bf16
                      const int* __restrict__ idx,
                      float* __restrict__ out,                   // [B,R] fp32
                      int t) {
    const int wave = threadIdx.x >> 5;
    const int lane = threadIdx.x & 31;
    const int m0 = blockIdx.x * 128 + (wave >> 1) * 32;   // 4 waves along M
    const int n0 = blockIdx.y * 32  + (wave & 1) * 16;    // 2 waves along N
    const int r  = lane & 15;

    v8f acc[2][5] = {};
    for (int k0 = 0; k0 < RN; k0 += 32) {
        FragA a0, a1;
        load_fragA(a0, h_in, m0 + r,      k0, lane);
        load_fragA(a1, h_in, m0 + 16 + r, k0, lane);
        #pragma unroll
        for (int g = 0; g < 5; ++g) {
            FragB b;
            load_fragB(b, WhT, g * RN + n0 + r, k0, lane);
            acc[0][g] = WMMA_BF16(a0.v, b.v, acc[0][g]);
            acc[1][g] = WMMA_BF16(a1.v, b.v, acc[1][g]);
        }
    }

    // Epilogue: C/D layout -> lane holds (m = mbase+v, n) for v = 0..7
    const int n   = n0 + (lane & 15);
    const int hi8 = (((lane >> 4) & 1) << 3);
    #pragma unroll
    for (int mt = 0; mt < 2; ++mt) {
        const int mbase = m0 + mt * 16 + hi8;
        #pragma unroll
        for (int v = 0; v < 8; ++v) {
            const int m = mbase + v;
            const float* sp = S_pre_t + (long)m * FIVR;
            float si = acc[mt][0][v] + sp[0 * RN + n];
            float sf = acc[mt][1][v] + sp[1 * RN + n];
            float so = acc[mt][2][v] + sp[2 * RN + n];
            float t1 = acc[mt][3][v] + sp[3 * RN + n];
            float t2 = acc[mt][4][v] + sp[4 * RN + n];
            float ig = sigmoidf_fast(si);
            float fg = sigmoidf_fast(sf);
            float og = sigmoidf_fast(so);
            float tr = fmaxf(t1, t2);
            const long pos = (long)m * RN + n;
            float cnew = fg * c[pos] + ig * tr;
            float hnew = og * tanhf_fast(cnew);
            c[pos]     = cnew;
            h_out[pos] = f2bf(hnew);
            if (idx[m] == t) out[pos] = hnew;
        }
    }
}

// ---------------------------------------------------------------------------
// Host-side launch
// ---------------------------------------------------------------------------
extern "C" void kernel_launch(void* const* d_in, const int* in_sizes, int n_in,
                              void* d_out, int out_size, void* d_ws, size_t ws_size,
                              hipStream_t stream) {
    const int*   seq       = (const int*)  d_in[0];   // [B,T]
    const float* seq_masks = (const float*)d_in[1];   // [B,T]
    const float* embed_w   = (const float*)d_in[2];   // [V+1,E]
    const float* Wi        = (const float*)d_in[3];   // [E,5R]
    const float* bi        = (const float*)d_in[4];   // [5R]
    const float* Wh        = (const float*)d_in[5];   // [R,5R]
    const float* bh        = (const float*)d_in[6];   // [5R]
    float*       out       = (float*)d_out;           // [B,R]

    // Workspace carve-up (all offsets 256B aligned)
    char* ws = (char*)d_ws;
    const size_t SZ_WT  = (size_t)FIVR * EN * sizeof(unsigned short); // 2.62 MB each
    const size_t SZ_X   = (size_t)GN * EN * sizeof(unsigned short);   // 41.9 MB
    const size_t SZ_S   = (size_t)GN * FIVR * sizeof(float);          // 419 MB
    const size_t SZ_H   = (size_t)BN * RN * sizeof(unsigned short);   // 1 MB
    const size_t SZ_C   = (size_t)BN * RN * sizeof(float);            // 2 MB
    unsigned short* WiT = (unsigned short*)(ws);
    unsigned short* WhT = (unsigned short*)(ws + SZ_WT);
    unsigned short* Xbf = (unsigned short*)(ws + 2 * SZ_WT);
    float*          Spre= (float*)         (ws + 2 * SZ_WT + SZ_X);
    unsigned short* h0  = (unsigned short*)(ws + 2 * SZ_WT + SZ_X + SZ_S);
    unsigned short* h1  = (unsigned short*)(ws + 2 * SZ_WT + SZ_X + SZ_S + SZ_H);
    float*          cst = (float*)         (ws + 2 * SZ_WT + SZ_X + SZ_S + 2 * SZ_H);
    int*            idx = (int*)           (ws + 2 * SZ_WT + SZ_X + SZ_S + 2 * SZ_H + SZ_C);

    // 1) gather index per batch row
    idx_kernel<<<(BN + 255) / 256, 256, 0, stream>>>(seq_masks, idx);

    // 2) weight transpose + bf16 convert
    {
        long total = (long)EN * FIVR;
        int blocks = (int)((total + 255) / 256);
        transpose_bf_kernel<<<blocks, 256, 0, stream>>>(Wi, WiT, EN, FIVR);
        transpose_bf_kernel<<<blocks, 256, 0, stream>>>(Wh, WhT, RN, FIVR);
    }

    // 3) zero h0, c
    init_state_kernel<<<(BN * RN + 255) / 256, 256, 0, stream>>>(h0, cst);

    // 4) embedding gather -> bf16 X
    embed_gather_kernel<<<(GN * EN / 8 + 255) / 256, 256, 0, stream>>>(seq, embed_w, Xbf);

    // 5) big GEMM: S_pre = X @ WiT^T + bi + bh   (wave tile 32x32)
    gemm1_kernel<<<dim3(GN / 64, FIVR / 128), 256, 0, stream>>>(Xbf, WiT, bi, bh, Spre);

    // 6) 40 sequential LSTM steps (ping-pong h)
    for (int t = 0; t < SN; ++t) {
        const unsigned short* hin  = (t & 1) ? h1 : h0;
        unsigned short*       hout = (t & 1) ? h0 : h1;
        const float* sp = Spre + (size_t)t * BN * FIVR;
        lstm_step_kernel<<<dim3(BN / 128, RN / 32), 256, 0, stream>>>(
            hin, WhT, sp, cst, hout, idx, out, t);
    }
}
